// Leader_75892072120402
// MI455X (gfx1250) — matrix operand order repaired and checked by reading, not compiled, not run on hardware
//
#include <hip/hip_runtime.h>
#include <hip/hip_bf16.h>

typedef __attribute__((ext_vector_type(16))) _Float16 v16h;
typedef __attribute__((ext_vector_type(8)))  _Float16 v8h;
typedef __attribute__((ext_vector_type(8)))  float    v8f;

#define WMMA(a,b,c) __builtin_amdgcn_wmma_f32_16x16x32_f16(false,(a),false,(b),(short)0,(c),false,false)

constexpr int BB  = 8192;
constexpr int TT  = 200;
constexpr int OBSN= 8;
constexpr int STN = 16;
constexpr int CTN = 4;

// Weight fragment table offsets (in fragments; one fragment = 512 halfs = 32 lanes x 16)
constexpr int F_ENCWI = 0;    // 12 frags: rnn_Wi  [192x8]  K pad 32
constexpr int F_ENCWH = 12;   // 24 frags: rnn_Wh  [192x64]
constexpr int F_RWI   = 36;   // 12 frags: rnnc_Wi [192x32]
constexpr int F_RWH   = 48;   // 24 frags: rnnc_Wh [192x64]
constexpr int F_MCW0  = 72;   // 12 frags: mc_W0   [64x80] K pad 96
constexpr int F_MCW   = 84;   // 24 frags: mc_W    [3][64x64]
constexpr int F_WOUT  = 108;  //  2 frags: mc_Wout [4x64]
constexpr int F_SYSC  = 110;  //  1 frag : sys_C   [8x16]  K pad 32
constexpr int F_SYSAB = 111;  //  1 frag : [A|B]   [16x20] K pad 32
constexpr int F_MLPW0 = 112;  // 12 frags: mlp_W0  [64x72] K pad 96
constexpr int F_MLPW1 = 124;  //  2 frags: mlp_W1  [16x64]
constexpr int F_TOT   = 126;

constexpr int WAVE_SCRATCH = 3584;  // halfs per wave: inpt 512 + ubuf 1536 + xbuf 1024 + sbuf 512
constexpr int SMEM_HALFS   = F_TOT * 512 + 4 * WAVE_SCRATCH;

__device__ __forceinline__ v8f splat8(float b) {
  v8f c;
#pragma unroll
  for (int i = 0; i < 8; i++) c[i] = b;
  return c;
}
// Fast sigmoid/tanh: v_exp_f32 + v_rcp_f32 (no IEEE divide expansion).
__device__ __forceinline__ float sigf(float x) {
  return __builtin_amdgcn_rcpf(1.0f + __expf(-x));
}
__device__ __forceinline__ float tanhfast(float x) { return 2.0f * sigf(2.0f * x) - 1.0f; }

// A-fragment (16xK tile, K-block at kbase) from a row-major f16 LDS buffer.
// A 16x32 layout (contiguous, matching B frags): lane = M (mod 16); elem e -> K = kbase + hi*16 + e.
__device__ __forceinline__ v16h ldsA(const _Float16* buf, int strideH, int m, int hi, int kbase) {
  const _Float16* p = buf + m * strideH + kbase + hi * 16;
  v8h lo = *(const v8h*)p;
  v8h hv = *(const v8h*)(p + 8);
  v16h r;
#pragma unroll
  for (int i = 0; i < 8; i++) { r[i] = lo[i]; r[i + 8] = hv[i]; }
  return r;
}

// B-fragment: pre-swizzled in LDS, 16 contiguous halfs per lane.
__device__ __forceinline__ v16h ldsB(const _Float16* wf, int frag, int lane) {
  return *(const v16h*)(wf + frag * 512 + lane * 16);
}

// Transpose-store a C/D f32 tile (lane holds col n, rows r+8*hi) into row-major f16 LDS.
__device__ __forceinline__ void storeT(_Float16* buf, int strideH, int nn, int hi, int colbase, v8f c) {
#pragma unroll
  for (int r = 0; r < 8; r++)
    buf[(r + 8 * hi) * strideH + colbase + nn] = (_Float16)c[r];
}

__device__ __forceinline__ v8h cvt8(float4 a, float4 b) {
  v8h o;
  o[0] = (_Float16)a.x; o[1] = (_Float16)a.y; o[2] = (_Float16)a.z; o[3] = (_Float16)a.w;
  o[4] = (_Float16)b.x; o[5] = (_Float16)b.y; o[6] = (_Float16)b.z; o[7] = (_Float16)b.w;
  return o;
}

// Fill B-fragments for W [OUTS x K] used as (x @ W.T): B[k][n] = W[tile*16+n][k].
// Frag layout: lane (n,hi), element e -> k = frag*32 + hi*16 + e. Zero-pad out of range.
__device__ void fillW(const float* W, int OUTS, int K, int ntiles, int nfrags, _Float16* dst) {
  int tot = ntiles * nfrags * 512;
  for (int idx = threadIdx.x; idx < tot; idx += blockDim.x) {
    int e  = idx & 15;
    int ln = (idx >> 4) & 31;
    int tf = idx >> 9;
    int o16 = ln & 15, hh = ln >> 4;
    int f = tf % nfrags, tl = tf / nfrags;
    int o = tl * 16 + o16;
    int k = f * 32 + hh * 16 + e;
    float v = (o < OUTS && k < K) ? W[o * K + k] : 0.0f;
    dst[tf * 512 + ln * 16 + e] = (_Float16)v;
  }
}

// Combined [A(16x16) | B(16x4)] for ns = state@A.T + cmd@B.T (single 16x32-padded frag).
__device__ void fillAB(const float* A, const float* Bm, _Float16* dst) {
  for (int idx = threadIdx.x; idx < 512; idx += blockDim.x) {
    int e = idx & 15; int ln = idx >> 4;
    int o = ln & 15, hh = ln >> 4;
    int k = hh * 16 + e;
    float v = 0.0f;
    if (k < 16)      v = A[o * 16 + k];
    else if (k < 20) v = Bm[o * 4 + (k - 16)];
    dst[ln * 16 + e] = (_Float16)v;
  }
}

__global__ void __launch_bounds__(128)
ctrl_rollout_kernel(
    const float* __restrict__ obs,
    const float* sysA, const float* sysB, const float* sysC,
    const float* rWi, const float* rWh, const float* rbi, const float* rbh,
    const float* cWi, const float* cWh, const float* cbi, const float* cbh,
    const float* mW0, const float* mb0, const float* mW1, const float* mb1,
    const float* mcW0, const float* mcb0, const float* lng_, const float* lnb_,
    const float* mcW, const float* mcb, const float* mcWo, const float* mcbo,
    float* __restrict__ out_obs, float* __restrict__ out_cmd, float* __restrict__ out_st)
{
  extern __shared__ char smem_raw[];
  _Float16* smem = (_Float16*)smem_raw;

  const int tid  = threadIdx.x;
  const int lane = tid & 31;
  const int wid  = tid >> 5;
  const int nn   = lane & 15;
  const int hi   = lane >> 4;
  const int bbase = blockIdx.x * 64 + wid * 16;

  _Float16* wf   = smem;                                    // weight fragments (shared)
  _Float16* inpt = smem + F_TOT * 512 + wid * WAVE_SCRATCH; // 16x32: [state|obs|err]
  _Float16* ubuf = inpt + 512;                              // 16x96: [hc(64)|obs(8)|err(8)|0(16)]
  _Float16* xbuf = ubuf + 1536;                             // 16x64
  _Float16* sbuf = xbuf + 1024;                             // 16x32: [state|cmd|0]

  // Zero this wave's scratch (ensures zero K-padding regions).
  {
    v8h z = {};
#pragma unroll
    for (int i = 0; i < WAVE_SCRATCH / (32 * 8); i++)
      *(v8h*)(inpt + (i * 32 + lane) * 8) = z;
  }

  // Convert + swizzle all weights into LDS B-fragments (block cooperative).
  fillW(rWi,  192,  8, 12, 1, wf + F_ENCWI * 512);
  fillW(rWh,  192, 64, 12, 2, wf + F_ENCWH * 512);
  fillW(cWi,  192, 32, 12, 1, wf + F_RWI   * 512);
  fillW(cWh,  192, 64, 12, 2, wf + F_RWH   * 512);
  fillW(mcW0,  64, 80,  4, 3, wf + F_MCW0  * 512);
#pragma unroll 1
  for (int l = 0; l < 3; l++)
    fillW(mcW + l * 64 * 64, 64, 64, 4, 2, wf + (F_MCW + l * 8) * 512);
  fillW(mcWo,   4, 64,  1, 2, wf + F_WOUT  * 512);
  fillW(sysC,   8, 16,  1, 1, wf + F_SYSC  * 512);
  fillAB(sysA, sysB,          wf + F_SYSAB * 512);
  fillW(mW0,   64, 72,  4, 3, wf + F_MLPW0 * 512);
  fillW(mW1,   16, 64,  1, 2, wf + F_MLPW1 * 512);
  __syncthreads();

  // ---- Per-lane bias preloads (column nn of each 16-wide output tile) ----
  float eb[8], ebin[4], ebhn[4], rb[8], rbin[4], rbhn[4];
#pragma unroll
  for (int j = 0; j < 8; j++) {
    eb[j] = rbi[16 * j + nn] + rbh[16 * j + nn];
    rb[j] = cbi[16 * j + nn] + cbh[16 * j + nn];
  }
#pragma unroll
  for (int j = 0; j < 4; j++) {
    ebin[j] = rbi[16 * (8 + j) + nn]; ebhn[j] = rbh[16 * (8 + j) + nn];
    rbin[j] = cbi[16 * (8 + j) + nn]; rbhn[j] = cbh[16 * (8 + j) + nn];
  }
  float mlpb0[4], mcb0v[4], lg[3][4], lb[3][4], mcbv[3][4];
#pragma unroll
  for (int t4 = 0; t4 < 4; t4++) { mlpb0[t4] = mb0[16 * t4 + nn]; mcb0v[t4] = mcb0[16 * t4 + nn]; }
#pragma unroll
  for (int l = 0; l < 3; l++)
#pragma unroll
    for (int t4 = 0; t4 < 4; t4++) {
      lg[l][t4]   = lng_[l * 64 + 16 * t4 + nn];
      lb[l][t4]   = lnb_[l * 64 + 16 * t4 + nn];
      mcbv[l][t4] = mcb[l * 64 + 16 * t4 + nn];
    }
  float mlpb1 = mb1[nn];
  float boutv = (nn < CTN) ? mcbo[nn] : 0.0f;

  // ================= Encoder GRU over T=200 =================
  v8f h[4];
#pragma unroll
  for (int j = 0; j < 4; j++) h[j] = splat8(0.0f);

#pragma unroll 1
  for (int t = 0; t < TT; t++) {
    // x A-fragment direct from global (K=8, rest zero-padded)
    v16h xa = {};
    if (hi == 0) {
      const float4* p = (const float4*)(obs + ((size_t)(bbase + nn) * TT + t) * OBSN);
      v8h oh = cvt8(p[0], p[1]);
#pragma unroll
      for (int i = 0; i < 8; i++) xa[i] = oh[i];
    }
    v16h a0 = ldsA(ubuf, 96, nn, hi, 0);
    v16h a1 = ldsA(ubuf, 96, nn, hi, 32);

    v8f g[8], gin[4], ghn[4];
#pragma unroll
    for (int j = 0; j < 8; j++) {
      v8f c = splat8(eb[j]);
      c = WMMA(xa, ldsB(wf, F_ENCWI + j, lane), c);
      c = WMMA(a0, ldsB(wf, F_ENCWH + 2 * j, lane), c);
      g[j] = WMMA(a1, ldsB(wf, F_ENCWH + 2 * j + 1, lane), c);
    }
#pragma unroll
    for (int j = 0; j < 4; j++) {
      gin[j] = WMMA(xa, ldsB(wf, F_ENCWI + 8 + j, lane), splat8(ebin[j]));
      v8f c = WMMA(a0, ldsB(wf, F_ENCWH + 2 * (8 + j), lane), splat8(ebhn[j]));
      ghn[j] = WMMA(a1, ldsB(wf, F_ENCWH + 2 * (8 + j) + 1, lane), c);
    }
#pragma unroll
    for (int j = 0; j < 4; j++) {
#pragma unroll
      for (int r = 0; r < 8; r++) {
        float rr = sigf(g[j][r]);
        float zz = sigf(g[4 + j][r]);
        float nv = tanhfast(gin[j][r] + rr * ghn[j][r]);
        h[j][r] = (1.0f - zz) * nv + zz * h[j][r];
      }
      storeT(ubuf, 96, nn, hi, 16 * j, h[j]);
    }
  }

  // ================= Init MLP -> initial_state =================
  if (hi == 0) {
    const float4* p = (const float4*)(obs + ((size_t)(bbase + nn) * TT + 0) * OBSN);
    *(v8h*)(ubuf + nn * 96 + 64) = cvt8(p[0], p[1]);   // cols 72..95 stay zero
  }
  {
    v16h u0 = ldsA(ubuf, 96, nn, hi, 0);
    v16h u1 = ldsA(ubuf, 96, nn, hi, 32);
    v16h u2 = ldsA(ubuf, 96, nn, hi, 64);
#pragma unroll
    for (int t4 = 0; t4 < 4; t4++) {
      v8f c = splat8(mlpb0[t4]);
      c = WMMA(u0, ldsB(wf, F_MLPW0 + 3 * t4, lane), c);
      c = WMMA(u1, ldsB(wf, F_MLPW0 + 3 * t4 + 1, lane), c);
      c = WMMA(u2, ldsB(wf, F_MLPW0 + 3 * t4 + 2, lane), c);
#pragma unroll
      for (int r = 0; r < 8; r++) c[r] = tanhfast(c[r]);
      storeT(xbuf, 64, nn, hi, 16 * t4, c);
    }
    v16h xa0 = ldsA(xbuf, 64, nn, hi, 0);
    v16h xa1 = ldsA(xbuf, 64, nn, hi, 32);
    v8f s = splat8(mlpb1);
    s = WMMA(xa0, ldsB(wf, F_MLPW1, lane), s);
    s = WMMA(xa1, ldsB(wf, F_MLPW1 + 1, lane), s);
#pragma unroll
    for (int r = 0; r < 8; r++)
      out_st[(size_t)(bbase + r + 8 * hi) * (TT * STN) + nn] = s[r];
    storeT(inpt, 32, nn, hi, 0, s);
    storeT(sbuf, 32, nn, hi, 0, s);
  }
  // zero controller hidden
#pragma unroll
  for (int j = 0; j < 4; j++) { h[j] = splat8(0.0f); storeT(ubuf, 96, nn, hi, 16 * j, h[j]); }

  // ================= Closed-loop rollout t = 1..199 =================
#pragma unroll 1
  for (int t = 1; t < TT; t++) {
    // yhat = state @ C.T   (C frag zero for k>=16, so stale obs/err cols are harmless)
    v16h ai = ldsA(inpt, 32, nn, hi, 0);
    v8f y = WMMA(ai, ldsB(wf, F_SYSC, lane), splat8(0.0f));
    if (nn < OBSN) {
#pragma unroll
      for (int r = 0; r < 8; r++)
        out_obs[(size_t)(bbase + r + 8 * hi) * (TT * OBSN) + (size_t)(t - 1) * OBSN + nn] = y[r];
    }
    // obs_t -> inpt[16..23] and ubuf[64..71]
    if (hi == 0) {
      const float4* p = (const float4*)(obs + ((size_t)(bbase + nn) * TT + t) * OBSN);
      v8h oh = cvt8(p[0], p[1]);
      *(v8h*)(inpt + nn * 32 + 16) = oh;
      *(v8h*)(ubuf + nn * 96 + 64) = oh;
    }
    // err = obs - yhat -> inpt[24..31] and ubuf[72..79]
    if (nn < OBSN) {
#pragma unroll
      for (int r = 0; r < 8; r++) {
        float ov = (float)inpt[(r + 8 * hi) * 32 + 16 + nn];
        _Float16 eh = (_Float16)(ov - y[r]);
        inpt[(r + 8 * hi) * 32 + 24 + nn] = eh;
        ubuf[(r + 8 * hi) * 96 + 72 + nn] = eh;
      }
    }
    // Controller GRU
    v16h an  = ldsA(inpt, 32, nn, hi, 0);
    v16h ah0 = ldsA(ubuf, 96, nn, hi, 0);
    v16h ah1 = ldsA(ubuf, 96, nn, hi, 32);
    v8f g[8], gin[4], ghn[4];
#pragma unroll
    for (int j = 0; j < 8; j++) {
      v8f c = splat8(rb[j]);
      c = WMMA(an,  ldsB(wf, F_RWI + j, lane), c);
      c = WMMA(ah0, ldsB(wf, F_RWH + 2 * j, lane), c);
      g[j] = WMMA(ah1, ldsB(wf, F_RWH + 2 * j + 1, lane), c);
    }
#pragma unroll
    for (int j = 0; j < 4; j++) {
      gin[j] = WMMA(an, ldsB(wf, F_RWI + 8 + j, lane), splat8(rbin[j]));
      v8f c = WMMA(ah0, ldsB(wf, F_RWH + 2 * (8 + j), lane), splat8(rbhn[j]));
      ghn[j] = WMMA(ah1, ldsB(wf, F_RWH + 2 * (8 + j) + 1, lane), c);
    }
#pragma unroll
    for (int j = 0; j < 4; j++) {
#pragma unroll
      for (int r = 0; r < 8; r++) {
        float rr = sigf(g[j][r]);
        float zz = sigf(g[4 + j][r]);
        float nv = tanhfast(gin[j][r] + rr * ghn[j][r]);
        h[j][r] = (1.0f - zz) * nv + zz * h[j][r];
      }
      storeT(ubuf, 96, nn, hi, 16 * j, h[j]);
    }
    // Control MLP layer 0: x = relu(u @ W0.T + b0), u = [hc|obs|err|0] K=96
    v16h u0 = ldsA(ubuf, 96, nn, hi, 0);
    v16h u1 = ldsA(ubuf, 96, nn, hi, 32);
    v16h u2 = ldsA(ubuf, 96, nn, hi, 64);
    v8f x4[4];
#pragma unroll
    for (int t4 = 0; t4 < 4; t4++) {
      v8f c = splat8(mcb0v[t4]);
      c = WMMA(u0, ldsB(wf, F_MCW0 + 3 * t4, lane), c);
      c = WMMA(u1, ldsB(wf, F_MCW0 + 3 * t4 + 1, lane), c);
      c = WMMA(u2, ldsB(wf, F_MCW0 + 3 * t4 + 2, lane), c);
#pragma unroll
      for (int r = 0; r < 8; r++) c[r] = fmaxf(c[r], 0.0f);
      x4[t4] = c;
    }
    // 3x (LayerNorm -> matmul -> relu)
#pragma unroll
    for (int l = 0; l < 3; l++) {
#pragma unroll
      for (int r = 0; r < 8; r++) {
        float sm = x4[0][r] + x4[1][r] + x4[2][r] + x4[3][r];
        float sq = x4[0][r] * x4[0][r] + x4[1][r] * x4[1][r]
                 + x4[2][r] * x4[2][r] + x4[3][r] * x4[3][r];
#pragma unroll
        for (int o = 1; o < 16; o <<= 1) {
          sm += __shfl_xor(sm, o, 32);
          sq += __shfl_xor(sq, o, 32);
        }
        float mean = sm * (1.0f / 64.0f);
        float var  = sq * (1.0f / 64.0f) - mean * mean;
        float rs   = __builtin_amdgcn_rsqf(var + 1e-5f);
#pragma unroll
        for (int t4 = 0; t4 < 4; t4++) {
          float xn = (x4[t4][r] - mean) * rs * lg[l][t4] + lb[l][t4];
          xbuf[(r + 8 * hi) * 64 + 16 * t4 + nn] = (_Float16)xn;
        }
      }
      v16h b0 = ldsA(xbuf, 64, nn, hi, 0);
      v16h b1 = ldsA(xbuf, 64, nn, hi, 32);
#pragma unroll
      for (int t4 = 0; t4 < 4; t4++) {
        v8f c = splat8(mcbv[l][t4]);
        c = WMMA(b0, ldsB(wf, F_MCW + l * 8 + 2 * t4, lane), c);
        c = WMMA(b1, ldsB(wf, F_MCW + l * 8 + 2 * t4 + 1, lane), c);
#pragma unroll
        for (int r = 0; r < 8; r++) c[r] = fmaxf(c[r], 0.0f);
        x4[t4] = c;
      }
    }
    // cmd = x @ Wout.T + bout
#pragma unroll
    for (int t4 = 0; t4 < 4; t4++) storeT(xbuf, 64, nn, hi, 16 * t4, x4[t4]);
    v16h c0 = ldsA(xbuf, 64, nn, hi, 0);
    v16h c1 = ldsA(xbuf, 64, nn, hi, 32);
    v8f cmd = splat8(boutv);
    cmd = WMMA(c0, ldsB(wf, F_WOUT, lane), cmd);
    cmd = WMMA(c1, ldsB(wf, F_WOUT + 1, lane), cmd);
    if (nn < CTN) {
#pragma unroll
      for (int r = 0; r < 8; r++) {
        out_cmd[(size_t)(bbase + r + 8 * hi) * ((TT - 1) * CTN) + (size_t)(t - 1) * CTN + nn] = cmd[r];
        sbuf[(r + 8 * hi) * 32 + 16 + nn] = (_Float16)cmd[r];
      }
    }
    // ns = state@A.T + cmd@B.T, clamp-mask, outputs, feed next step
    v16h as = ldsA(sbuf, 32, nn, hi, 0);
    v8f ns = WMMA(as, ldsB(wf, F_SYSAB, lane), splat8(0.0f));
#pragma unroll
    for (int r = 0; r < 8; r++) {
      float v = ns[r];
      ns[r] = (v > -100.0f && v < 100.0f) ? v : 0.0f;
      out_st[(size_t)(bbase + r + 8 * hi) * (TT * STN) + (size_t)t * STN + nn] = ns[r];
    }
    storeT(inpt, 32, nn, hi, 0, ns);
    storeT(sbuf, 32, nn, hi, 0, ns);
  }

  // Final observation_hat[:, 199]
  {
    v16h af = ldsA(inpt, 32, nn, hi, 0);
    v8f yf = WMMA(af, ldsB(wf, F_SYSC, lane), splat8(0.0f));
    if (nn < OBSN) {
#pragma unroll
      for (int r = 0; r < 8; r++)
        out_obs[(size_t)(bbase + r + 8 * hi) * (TT * OBSN) + (size_t)(TT - 1) * OBSN + nn] = yf[r];
    }
  }
}

extern "C" void kernel_launch(void* const* d_in, const int* in_sizes, int n_in,
                              void* d_out, int out_size, void* d_ws, size_t ws_size,
                              hipStream_t stream) {
  (void)in_sizes; (void)n_in; (void)out_size; (void)d_ws; (void)ws_size;
  const float* obs  = (const float*)d_in[0];
  const float* sysA = (const float*)d_in[1];
  const float* sysB = (const float*)d_in[2];
  const float* sysC = (const float*)d_in[3];
  const float* rWi  = (const float*)d_in[4];
  const float* rWh  = (const float*)d_in[5];
  const float* rbi  = (const float*)d_in[6];
  const float* rbh  = (const float*)d_in[7];
  const float* cWi  = (const float*)d_in[8];
  const float* cWh  = (const float*)d_in[9];
  const float* cbi  = (const float*)d_in[10];
  const float* cbh  = (const float*)d_in[11];
  const float* mW0  = (const float*)d_in[12];
  const float* mb0  = (const float*)d_in[13];
  const float* mW1  = (const float*)d_in[14];
  const float* mb1  = (const float*)d_in[15];
  const float* mcW0 = (const float*)d_in[16];
  const float* mcb0 = (const float*)d_in[17];
  const float* lng_ = (const float*)d_in[18];
  const float* lnb_ = (const float*)d_in[19];
  const float* mcW  = (const float*)d_in[20];
  const float* mcb  = (const float*)d_in[21];
  const float* mcWo = (const float*)d_in[22];
  const float* mcbo = (const float*)d_in[23];

  float* out_obs = (float*)d_out;
  float* out_cmd = out_obs + (size_t)BB * TT * OBSN;          // 13,107,200
  float* out_st  = out_cmd + (size_t)BB * (TT - 1) * CTN;     // + 6,520,832

  const size_t smem_bytes = (size_t)SMEM_HALFS * sizeof(short); // ~157.7 KB (< 320 KB WGP LDS)
  (void)hipFuncSetAttribute((const void*)ctrl_rollout_kernel,
                            hipFuncAttributeMaxDynamicSharedMemorySize,
                            (int)smem_bytes);

  dim3 grid(BB / 64);   // 128 blocks, 64 batch rows each (4 waves x 16 rows)
  dim3 block(128);
  ctrl_rollout_kernel<<<grid, block, smem_bytes, stream>>>(
      obs, sysA, sysB, sysC, rWi, rWh, rbi, rbh, cWi, cWh, cbi, cbh,
      mW0, mb0, mW1, mb1, mcW0, mcb0, lng_, lnb_, mcW, mcb, mcWo, mcbo,
      out_obs, out_cmd, out_st);
}
